// PARAttention_31825707664131
// MI455X (gfx1250) — compile-verified, matrix operand order checked by reading, not run on hardware
//
#include <hip/hip_runtime.h>

// ---------------------------------------------------------------------------
// PARAttention on gfx1250 (CDNA5): bf16 WMMA pipeline.
//   x(16,8256,512) -> kv GEMM + RoPE + LN, q GEMM (latent rows only) + RoPE,
//   flash attention (64 latent q x 8256 kv per (b,h)), combine.
// All matrix math uses v_wmma_f32_16x16x32_bf16 (wave32).
// Cross-lane reductions use DPP (quad_perm/row_mirror); softmax row-sums are
// computed by an extra WMMA against a ones-column. Attention does a
// single-pass (non-online) softmax per 192-key chunk: 12 sim accumulator
// tiles are held in registers, one max reduction per chunk, no rescaling.
// ---------------------------------------------------------------------------

typedef __bf16 bf16;
typedef __attribute__((ext_vector_type(16))) __bf16 v16bf;
typedef __attribute__((ext_vector_type(8)))  float  v8f;

constexpr int   BATCH = 16;
constexpr int   SEQ   = 8192;
constexpr int   LAT   = 64;
constexpr int   NPOS  = SEQ + LAT;      // 8256
constexpr int   DIMC  = 512;
constexpr int   DH    = 64;
constexpr int   NHEAD = 8;
constexpr int   NBH   = BATCH * NHEAD;  // 128
constexpr float LNEPS = 1e-5f;
constexpr int   JSPLIT = 43;            // 8256 = 43 * 192
constexpr int   JCHUNK = NPOS / JSPLIT; // 192 keys = 12 sim tiles
constexpr int   NTILE  = JCHUNK / 16;   // 12
constexpr int   PSTR   = 200;           // LDS row stride (halves), 16B aligned

// workspace layout (bytes)
constexpr size_t OFF_XB   = 0;                                   // x bf16
constexpr size_t OFF_WQ   = OFF_XB   + (size_t)BATCH*NPOS*DIMC*2;
constexpr size_t OFF_WKV  = OFF_WQ   + (size_t)DIMC*DIMC*2;
constexpr size_t OFF_QB   = OFF_WKV  + (size_t)DIMC*DIMC*2;      // q[bh][i][d]
constexpr size_t OFF_KVR  = OFF_QB   + (size_t)NBH*LAT*DH*2;     // kv_rope[bh][pos][d]
constexpr size_t OFF_LKVT = OFF_KVR  + (size_t)NBH*NPOS*DH*2;    // lkvT[bh][d][pos]
constexpr size_t OFF_PO   = OFF_LKVT + (size_t)NBH*NPOS*DH*2;    // partial O f32
constexpr size_t OFF_PM   = OFF_PO   + (size_t)JSPLIT*NBH*LAT*DH*4;
constexpr size_t OFF_PL   = OFF_PM   + (size_t)JSPLIT*NBH*LAT*4;

__device__ inline bf16 f2bf(float f) { return (bf16)f; }

__device__ inline v8f zero8() {
  v8f z; for (int i = 0; i < 8; ++i) z[i] = 0.f; return z;
}

// ---- DPP 16-lane butterfly reductions (halves 0-15 / 16-31 stay separate) --
template <int CTRL>
__device__ inline float dppf(float x) {
  return __int_as_float(__builtin_amdgcn_update_dpp(
      0, __float_as_int(x), CTRL, 0xF, 0xF, true));
}
__device__ inline float red16_max(float x) {
  x = fmaxf(x, dppf<0xB1>(x));    // quad_perm(1,0,3,2)  : xor 1
  x = fmaxf(x, dppf<0x4E>(x));    // quad_perm(2,3,0,1)  : xor 2
  x = fmaxf(x, dppf<0x141>(x));   // row_half_mirror     : xor 7
  x = fmaxf(x, dppf<0x140>(x));   // row_mirror          : xor 15
  return x;
}
__device__ inline float red16_sum(float x) {
  x += dppf<0xB1>(x);
  x += dppf<0x4E>(x);
  x += dppf<0x141>(x);
  x += dppf<0x140>(x);
  return x;
}

// Load a 16-lane-striped WMMA fragment: lane holds 16 bf16 of one row,
// K-interleave {kk+0..7, kk+16..23} (lanes 0-15) / {kk+8..15, kk+24..31}
// (lanes 16-31). Two contiguous 16B loads per lane.
__device__ inline v16bf load_frag16(const bf16* __restrict__ row, int kk, int hi) {
  union { uint4 u[2]; v16bf v; } r;
  r.u[0] = *reinterpret_cast<const uint4*>(row + kk + hi * 8);
  r.u[1] = *reinterpret_cast<const uint4*>(row + kk + 16 + hi * 8);
  return r.v;
}

__device__ inline v8f wmma_bf16(v16bf a, v16bf b, v8f c) {
  return __builtin_amdgcn_wmma_f32_16x16x32_bf16(false, a, false, b,
                                                 (short)0, c, false, false);
}

// ---------------------------------------------------------------------------
// f32 -> bf16 bulk conversion (8 elements / thread)
// ---------------------------------------------------------------------------
__global__ void __launch_bounds__(256)
cvt_f32_bf16(const float* __restrict__ in, bf16* __restrict__ out, int n8) {
  int idx = blockIdx.x * 256 + threadIdx.x;
  if (idx >= n8) return;
  const float4* ip = reinterpret_cast<const float4*>(in) + (size_t)idx * 2;
  float4 a = ip[0], b = ip[1];
  union { __bf16 h[8]; uint4 u; } pk;
  pk.h[0] = f2bf(a.x); pk.h[1] = f2bf(a.y); pk.h[2] = f2bf(a.z); pk.h[3] = f2bf(a.w);
  pk.h[4] = f2bf(b.x); pk.h[5] = f2bf(b.y); pk.h[6] = f2bf(b.z); pk.h[7] = f2bf(b.w);
  reinterpret_cast<uint4*>(out)[idx] = pk.u;
}

// ---------------------------------------------------------------------------
// GEMM (x @ W^T) with fused RoPE (+ LayerNorm for kv, + scale for q).
// One wave -> 32 rows x 64 cols (one head): 2 A frags share each B frag,
// 8 WMMA per 32-wide K step.
//   IS_Q=false: write kv_rope[bh][pos][d] and lkvT[bh][d][pos] (bf16)
//   IS_Q=true : write q[bh][i][d] (bf16, pre-scaled by d^-0.5)
// ---------------------------------------------------------------------------
template <bool IS_Q>
__global__ void __launch_bounds__(128)
gemm_rope_ln(const bf16* __restrict__ xb, const bf16* __restrict__ wb,
             const float* __restrict__ lng, const float* __restrict__ lnb,
             bf16* __restrict__ kvr, bf16* __restrict__ lkvT,
             bf16* __restrict__ qo) {
  const int lane = threadIdx.x & 31;
  const int wave = threadIdx.x >> 5;
  const int p  = lane & 15;
  const int hi = lane >> 4;
  const int mtile = blockIdx.x * 4 + wave;   // 32-row tile
  const int h  = blockIdx.y;                 // head == 64-wide N tile
  const int n0 = h * DH;
  const int m0 = mtile * 32;

  // A operand rows for this lane (two 16-row subtiles)
  const bf16* arow[2];
#pragma unroll
  for (int mg = 0; mg < 2; ++mg) {
    int g = m0 + 16 * mg + p;
    long xrow;
    if constexpr (IS_Q) {
      int b = g >> 6, i = g & 63;
      xrow = (long)b * NPOS + SEQ + i;
    } else {
      xrow = g;
    }
    arow[mg] = xb + xrow * DIMC;
  }

  v8f C[2][4];
#pragma unroll
  for (int mg = 0; mg < 2; ++mg)
#pragma unroll
    for (int t = 0; t < 4; ++t) C[mg][t] = zero8();

  for (int kk = 0; kk < DIMC; kk += 32) {
    v16bf a0 = load_frag16(arow[0], kk, hi);
    v16bf a1 = load_frag16(arow[1], kk, hi);
#pragma unroll
    for (int t = 0; t < 4; ++t) {
      const bf16* brow = wb + (long)(n0 + 16 * t + p) * DIMC;  // W row = out feat
      v16bf bf = load_frag16(brow, kk, hi);
      C[0][t] = wmma_bf16(a0, bf, C[0][t]);
      C[1][t] = wmma_bf16(a1, bf, C[1][t]);
    }
  }

  // ---- epilogue: RoPE. Pair c <-> c+32 lives in (C0,C2)/(C1,C3) at the same
  // lane; inv_freq depends only on c%32 so cos/sin are shared per pair.
  const float kLF = 0.28782310585f;  // ln(10000)/32
  const float invf0 = __expf(-kLF * (float)p);
  const float invf1 = __expf(-kLF * (float)(16 + p));

#pragma unroll
  for (int mg = 0; mg < 2; ++mg) {
    const int ms = m0 + 16 * mg;
    int b_idx, posbase;
    if constexpr (IS_Q) {
      b_idx = ms >> 6;
      posbase = SEQ + (ms & 63) + hi * 8;
    } else {
      b_idx = ms / NPOS;
      posbase = (ms % NPOS) + hi * 8;
    }
    const int bh = b_idx * NHEAD + h;

    float r[4][8];
#pragma unroll
    for (int v = 0; v < 8; ++v) {
      float posf = (float)(posbase + v);
      float a0 = posf * invf0, a1 = posf * invf1;
      float c0 = __cosf(a0), s0 = __sinf(a0);
      float c1 = __cosf(a1), s1 = __sinf(a1);
      float x0 = C[mg][0][v], x1 = C[mg][1][v];
      float x2 = C[mg][2][v], x3 = C[mg][3][v];
      r[0][v] = x0 * c0 - x2 * s0;
      r[2][v] = x2 * c0 + x0 * s0;
      r[1][v] = x1 * c1 - x3 * s1;
      r[3][v] = x3 * c1 + x1 * s1;
    }

    if constexpr (IS_Q) {
#pragma unroll
      for (int t = 0; t < 4; ++t)
#pragma unroll
        for (int v = 0; v < 8; ++v) {
          int i = (ms & 63) + hi * 8 + v;
          qo[((long)bh * LAT + i) * DH + 16 * t + p] = f2bf(r[t][v] * 0.125f);
        }
    } else {
      // LayerNorm over d=64: lane partials over 4 accums + DPP 16-lane sum.
      float mu[8], rstd[8];
#pragma unroll
      for (int v = 0; v < 8; ++v) {
        float s  = r[0][v] + r[1][v] + r[2][v] + r[3][v];
        float q2 = r[0][v] * r[0][v] + r[1][v] * r[1][v] +
                   r[2][v] * r[2][v] + r[3][v] * r[3][v];
        s = red16_sum(s);
        q2 = red16_sum(q2);
        float m_ = s * (1.f / 64.f);
        float var = q2 * (1.f / 64.f) - m_ * m_;
        mu[v] = m_;
        rstd[v] = rsqrtf(var + LNEPS);
      }
#pragma unroll
      for (int t = 0; t < 4; ++t) {
        int c = 16 * t + p;
        float gt = lng[c], bt = lnb[c];
        // K operand for sim: kv_rope[bh][pos][d]
#pragma unroll
        for (int v = 0; v < 8; ++v)
          kvr[(long)bh * NPOS * DH + (long)(posbase + v) * DH + c] =
              f2bf(r[t][v]);
        // V operand (transposed): lkvT[bh][d][pos]; 8 consecutive pos -> b128
        union { __bf16 hv[8]; uint4 u; } pk;
#pragma unroll
        for (int v = 0; v < 8; ++v)
          pk.hv[v] = f2bf((r[t][v] - mu[v]) * rstd[v] * gt + bt);
        *reinterpret_cast<uint4*>(lkvT + ((long)bh * DH + c) * NPOS + posbase) =
            pk.u;
      }
    }
  }
}

// ---------------------------------------------------------------------------
// Flash-attention partials, single-pass per chunk.
// grid = (JSPLIT=43, 128 bh), 4 waves/block; wave w owns query rows
// [16w,16w+16) and all 192 keys of its chunk:
//   phase 1: 12 sim tiles in registers (24 WMMA)
//   phase 2: one row-max per chunk (max3 chain + 4-step DPP butterfly)
//   phase 3: exp -> bf16 -> LDS (C-layout -> A-layout transpose)
//   phase 4: O = P x V (24 WMMA) + row sums l = P x ones (6 WMMA)
// No online rescaling needed; chunk max is exact, combine merges chunks.
// ---------------------------------------------------------------------------
__global__ void __launch_bounds__(128)
attn_partial(const bf16* __restrict__ qb, const bf16* __restrict__ kvr,
             const bf16* __restrict__ lkvT, float* __restrict__ pO,
             float* __restrict__ pm, float* __restrict__ pl) {
  __shared__ __bf16 sP[4 * 16 * PSTR];  // per-wave 16x192 P tile (padded rows)
  const int lane = threadIdx.x & 31;
  const int wave = threadIdx.x >> 5;
  const int p  = lane & 15;
  const int hi = lane >> 4;
  const int bh = blockIdx.y;
  const int chunk = blockIdx.x;
  const int i0 = wave * 16;
  const int jbase = chunk * JCHUNK;
  __bf16* wP = sP + wave * 16 * PSTR;

  const bf16* qrow = qb + ((long)bh * LAT + i0 + p) * DH;
  v16bf qA0 = load_frag16(qrow, 0, hi);
  v16bf qA1 = load_frag16(qrow, 32, hi);

  // phase 1: sim tiles over all 192 keys (K = d = 64 -> 2 WMMA each)
  v8f Cs[NTILE];
#pragma unroll
  for (int s = 0; s < NTILE; ++s) {
    const bf16* krow = kvr + ((long)bh * NPOS + jbase + 16 * s + p) * DH;
    v16bf B0 = load_frag16(krow, 0, hi);
    v16bf B1 = load_frag16(krow, 32, hi);
    v8f c = zero8();
    c = wmma_bf16(qA0, B0, c);
    c = wmma_bf16(qA1, B1, c);
    Cs[s] = c;
  }

  // phase 2: per-row chunk max (lanewise chain -> DPP 16-lane butterfly)
  float mx[8];
#pragma unroll
  for (int v = 0; v < 8; ++v) {
    float m = Cs[0][v];
#pragma unroll
    for (int s = 1; s < NTILE; ++s) m = fmaxf(m, Cs[s][v]);
    mx[v] = red16_max(m);
  }

  // phase 3: P = exp(sim - max) staged to LDS as bf16
#pragma unroll
  for (int v = 0; v < 8; ++v) {
    int row = v + hi * 8;
#pragma unroll
    for (int s = 0; s < NTILE; ++s)
      wP[row * PSTR + 16 * s + p] = f2bf(__expf(Cs[s][v] - mx[v]));
  }
  asm volatile("s_wait_dscnt 0x0" ::: "memory");  // wave-private LDS RAW

  // ones-column B fragment: column 0 (lane p==0) = 1.0, rest 0 -> row sums.
  v16bf onesf;
  {
    __bf16 o = (p == 0) ? (__bf16)1.f : (__bf16)0.f;
#pragma unroll
    for (int i = 0; i < 16; ++i) onesf[i] = o;
  }

  // phase 4: O = P x V, l = P x ones, over 6 x 32-key steps
  v8f O[4];
#pragma unroll
  for (int t = 0; t < 4; ++t) O[t] = zero8();
  v8f Ol = zero8();               // row sums l (valid at lanes p==0)

#pragma unroll
  for (int jt = 0; jt < JCHUNK / 32; ++jt) {
    union { uint4 u[2]; v16bf v; } pa;
    pa.u[0] = *reinterpret_cast<const uint4*>(wP + p * PSTR + 32 * jt + hi * 8);
    pa.u[1] = *reinterpret_cast<const uint4*>(wP + p * PSTR + 32 * jt + 16 + hi * 8);
#pragma unroll
    for (int t = 0; t < 4; ++t) {
      const bf16* vrow = lkvT + ((long)bh * DH + 16 * t + p) * NPOS;
      v16bf Bv = load_frag16(vrow, jbase + 32 * jt, hi);
      O[t] = wmma_bf16(pa.v, Bv, O[t]);
    }
    Ol = wmma_bf16(pa.v, onesf, Ol);
  }

  long base = ((long)chunk * NBH + bh) * LAT;
#pragma unroll
  for (int t = 0; t < 4; ++t)
#pragma unroll
    for (int v = 0; v < 8; ++v)
      pO[(base + i0 + v + hi * 8) * DH + 16 * t + p] = O[t][v];
  if (p == 0) {
#pragma unroll
    for (int v = 0; v < 8; ++v) {
      pm[base + i0 + v + hi * 8] = mx[v];
      pl[base + i0 + v + hi * 8] = Ol[v];
    }
  }
}

// ---------------------------------------------------------------------------
// Combine JSPLIT partials, write out[b][i][h*64+d] (f32).
// ---------------------------------------------------------------------------
__global__ void __launch_bounds__(512)
attn_combine(const float* __restrict__ pO, const float* __restrict__ pm,
             const float* __restrict__ pl, float* __restrict__ out) {
  int b = blockIdx.x >> 6;   // gridDim.x = 16*64
  int i = blockIdx.x & 63;
  int t = threadIdx.x;       // 0..511 = h*64+d
  int h = t >> 6, d = t & 63;
  int bh = b * NHEAD + h;
  float M = -1e30f;
  for (int c = 0; c < JSPLIT; ++c)
    M = fmaxf(M, pm[((long)c * NBH + bh) * LAT + i]);
  float den = 0.f, num = 0.f;
  for (int c = 0; c < JSPLIT; ++c) {
    long rb = ((long)c * NBH + bh) * LAT + i;
    float f = __expf(pm[rb] - M);
    den += pl[rb] * f;
    num += pO[rb * DH + d] * f;
  }
  out[((long)b * LAT + i) * DIMC + t] = num / den;
}

// ---------------------------------------------------------------------------
extern "C" void kernel_launch(void* const* d_in, const int* in_sizes, int n_in,
                              void* d_out, int out_size, void* d_ws, size_t ws_size,
                              hipStream_t stream) {
  const float* x   = (const float*)d_in[0];
  const float* wq  = (const float*)d_in[1];
  const float* wkv = (const float*)d_in[2];
  const float* lng = (const float*)d_in[3];
  const float* lnb = (const float*)d_in[4];
  float* out = (float*)d_out;

  char* ws = (char*)d_ws;
  bf16* xb   = (bf16*)(ws + OFF_XB);
  bf16* wqb  = (bf16*)(ws + OFF_WQ);
  bf16* wkvb = (bf16*)(ws + OFF_WKV);
  bf16* qb   = (bf16*)(ws + OFF_QB);
  bf16* kvr  = (bf16*)(ws + OFF_KVR);
  bf16* lkvT = (bf16*)(ws + OFF_LKVT);
  float* pO  = (float*)(ws + OFF_PO);
  float* pm  = (float*)(ws + OFF_PM);
  float* pl  = (float*)(ws + OFF_PL);

  // 1) bf16 conversions
  {
    int n8 = BATCH * NPOS * DIMC / 8;
    cvt_f32_bf16<<<(n8 + 255) / 256, 256, 0, stream>>>(x, xb, n8);
    int w8 = DIMC * DIMC / 8;
    cvt_f32_bf16<<<(w8 + 255) / 256, 256, 0, stream>>>(wq, wqb, w8);
    cvt_f32_bf16<<<(w8 + 255) / 256, 256, 0, stream>>>(wkv, wkvb, w8);
  }
  // 2) kv GEMM + RoPE + LN   (4128 32-row M-tiles, 8 head tiles)
  gemm_rope_ln<false><<<dim3(BATCH * NPOS / 32 / 4, NHEAD), 128, 0, stream>>>(
      xb, wkvb, lng, lnb, kvr, lkvT, nullptr);
  // 3) q GEMM (latent rows only) + RoPE + scale   (32 32-row M-tiles)
  gemm_rope_ln<true><<<dim3(BATCH * LAT / 32 / 4, NHEAD), 128, 0, stream>>>(
      xb, wqb, lng, lnb, nullptr, nullptr, qb);
  // 4) attention partials (single-pass softmax per chunk)
  attn_partial<<<dim3(JSPLIT, NBH), 128, 0, stream>>>(qb, kvr, lkvT, pO, pm, pl);
  // 5) combine + final layout
  attn_combine<<<BATCH * LAT, 512, 0, stream>>>(pO, pm, pl, out);
}
